// Encoder_1941325218388
// MI455X (gfx1250) — compile-verified, matrix operand order checked by reading, not run on hardware
//
#include <hip/hip_runtime.h>

typedef __attribute__((ext_vector_type(16))) _Float16 v16h;
typedef __attribute__((ext_vector_type(8)))  float    v8f;

// ---------------- constants ----------------
constexpr int NB    = 4;      // clouds
constexpr int NPER  = 1024;   // points per cloud
constexpr int NPTS  = NB * NPER;
constexpr int KNBR  = 64;     // max neighbors

// ---------------- helpers ----------------
__device__ __forceinline__ unsigned fenc(float f) {
  unsigned u = __float_as_uint(f);
  return (u >> 31) ? ~u : (u | 0x80000000u);
}
__device__ __forceinline__ float fdec(unsigned u) {
  return __uint_as_float((u >> 31) ? (u ^ 0x80000000u) : ~u);
}

// =====================================================================
// Kernel: radius-KNN via per-point bitonic sort of masked distances.
// One block (256 thr) per query point; LDS sort of 1024 (key,val) pairs.
// Writes global neighbor index or -1 (invalid).
// =====================================================================
__global__ __launch_bounds__(256) void knn_kernel(const float* __restrict__ pos,
                                                  float r2, int* __restrict__ nbr) {
  __shared__ float key[NPER];
  __shared__ int   val[NPER];
  const int i    = blockIdx.x;            // global point id
  const int base = (i >> 10) << 10;       // cloud base
  const float px = pos[i * 3 + 0], py = pos[i * 3 + 1], pz = pos[i * 3 + 2];

  for (int j = threadIdx.x; j < NPER; j += blockDim.x) {
    int g = base + j;
    float dx = pos[g * 3 + 0] - px;
    float dy = pos[g * 3 + 1] - py;
    float dz = pos[g * 3 + 2] - pz;
    float d2 = dx * dx + dy * dy + dz * dz;
    key[j] = (d2 <= r2) ? d2 : __builtin_inff();
    val[j] = g;
  }
  __syncthreads();

  // bitonic sort ascending (1024 elements, 256 threads)
  for (int k = 2; k <= NPER; k <<= 1) {
    for (int j = k >> 1; j > 0; j >>= 1) {
      for (int idx = threadIdx.x; idx < NPER; idx += blockDim.x) {
        int ixj = idx ^ j;
        if (ixj > idx) {
          bool up = ((idx & k) == 0);
          float a = key[idx], b = key[ixj];
          bool sw = up ? (a > b) : (a < b);
          if (sw) {
            key[idx] = b; key[ixj] = a;
            int tv = val[idx]; val[idx] = val[ixj]; val[ixj] = tv;
          }
        }
      }
      __syncthreads();
    }
  }
  if (threadIdx.x < KNBR) {
    float kk = key[threadIdx.x];
    nbr[i * KNBR + threadIdx.x] = (kk < __builtin_inff()) ? val[threadIdx.x] : -1;
  }
}

// =====================================================================
// Kernel: pack fp32 weight matrix [Kreal x N] into per-lane WMMA B
// fragments (f16), padded K to multiples of 32.
// packed[((kt*NT+nt)*32 + lane)*16 + e] with layout from CDNA5 ISA
// =====================================================================
__global__ void pack_weights(const float* __restrict__ W, unsigned short* __restrict__ out,
                             int Kreal, int N, int NT, int total) {
  int gid = blockIdx.x * blockDim.x + threadIdx.x;
  if (gid >= total) return;
  int e    = gid & 15;
  int lane = (gid >> 4) & 31;
  int t    = gid >> 9;
  int nt   = t % NT;
  int kt   = t / NT;
  int v = e >> 1, h = e & 1;
  int Kl = ((v >= 4) ? 16 : 0) + ((lane >= 16) ? 8 : 0) + (v & 3) * 2 + h;
  int Nl = lane & 15;
  int K  = kt * 32 + Kl;
  float valf = (K < Kreal) ? W[K * N + nt * 16 + Nl] : 0.0f;
  union { _Float16 h16; unsigned short u; } cv;
  cv.h16 = (_Float16)valf;
  out[gid] = cv.u;
}

// =====================================================================
// Device: one WMMA MLP layer, activations LDS(f16, row-major MxK) ->
// LDS(f16, row-major MxN). Bias + optional ReLU. 8 waves / block.
// =====================================================================
template <int M, int K, int N, bool RELU>
__device__ __forceinline__ void mlp_layer(const _Float16* actIn,
                                          const unsigned short* Wp,
                                          const float* bias,
                                          _Float16* actOut) {
  constexpr int MT = M / 16, NT = N / 16, KT = K / 32, NW = 8;
  const int lane = threadIdx.x & 31;
  const int wave = threadIdx.x >> 5;
  const int r16  = lane & 15;
  const int hi   = lane >> 4;
  const uint32_t* a32 = (const uint32_t*)actIn;
  for (int nt = wave; nt < NT; nt += NW) {
    for (int mt = 0; mt < MT; ++mt) {
      v8f acc = {};
      for (int kt = 0; kt < KT; ++kt) {
        union { v16h v; uint32_t u[8]; } A;
        int baseE = (mt * 16 + r16) * K + kt * 32 + hi * 8;  // element index (even)
#pragma unroll
        for (int q = 0; q < 4; ++q) {
          A.u[q]     = a32[(baseE >> 1) + q];
          A.u[4 + q] = a32[((baseE + 16) >> 1) + q];
        }
        v16h B = *(const v16h*)((const _Float16*)Wp +
                                ((size_t)(kt * NT + nt) * 32 + lane) * 16);
        acc = __builtin_amdgcn_wmma_f32_16x16x32_f16(false, A.v, false, B,
                                                     (short)0, acc, false, false);
      }
#pragma unroll
      for (int r = 0; r < 8; ++r) {
        int row = mt * 16 + r + hi * 8;
        int col = nt * 16 + r16;
        float o = acc[r] + bias[col];
        if (RELU) o = fmaxf(o, 0.0f);
        actOut[row * N + col] = (_Float16)o;
      }
    }
  }
}

// Final layer: no store of activations; masked column-max into LDS via
// ordered-uint atomicMax (-> ds_max_u32).
template <int M, int K, int N, bool MASK>
__device__ __forceinline__ void mlp_layer_max(const _Float16* actIn,
                                              const unsigned short* Wp,
                                              const float* bias,
                                              const int* validL,
                                              unsigned* colmax) {
  constexpr int MT = M / 16, NT = N / 16, KT = K / 32, NW = 8;
  const int lane = threadIdx.x & 31;
  const int wave = threadIdx.x >> 5;
  const int r16  = lane & 15;
  const int hi   = lane >> 4;
  const uint32_t* a32 = (const uint32_t*)actIn;
  for (int nt = wave; nt < NT; nt += NW) {
    for (int mt = 0; mt < MT; ++mt) {
      v8f acc = {};
      for (int kt = 0; kt < KT; ++kt) {
        union { v16h v; uint32_t u[8]; } A;
        int baseE = (mt * 16 + r16) * K + kt * 32 + hi * 8;
#pragma unroll
        for (int q = 0; q < 4; ++q) {
          A.u[q]     = a32[(baseE >> 1) + q];
          A.u[4 + q] = a32[((baseE + 16) >> 1) + q];
        }
        v16h B = *(const v16h*)((const _Float16*)Wp +
                                ((size_t)(kt * NT + nt) * 32 + lane) * 16);
        acc = __builtin_amdgcn_wmma_f32_16x16x32_f16(false, A.v, false, B,
                                                     (short)0, acc, false, false);
      }
#pragma unroll
      for (int r = 0; r < 8; ++r) {
        int row = mt * 16 + r + hi * 8;
        int col = nt * 16 + r16;
        float o = acc[r] + bias[col];
        bool ok = MASK ? (validL[row] != 0) : true;
        if (ok) atomicMax(&colmax[col], fenc(o));
      }
    }
  }
}

// =====================================================================
// Kernel: fused SA module. One block (256 thr) per query point.
// Gather 64 neighbor edge features -> LDS f16 -> 3 WMMA layers ->
// masked column max -> out[i, N3].
// =====================================================================
template <int FIN, int KPAD, int N1, int N2, int N3>
__global__ __launch_bounds__(256) void sa_kernel(const float* __restrict__ xin,
                                                 const float* __restrict__ pos,
                                                 const int* __restrict__ nbr,
                                                 const unsigned short* W1p, const float* b1,
                                                 const unsigned short* W2p, const float* b2,
                                                 const unsigned short* W3p, const float* b3,
                                                 float* __restrict__ out) {
  constexpr int M    = 64;
  constexpr int ABUF = M * ((KPAD > N2) ? KPAD : N2);
  __shared__ __align__(16) _Float16 bufA[ABUF];
  __shared__ __align__(16) _Float16 bufB[M * N1];
  __shared__ unsigned colmax[N3];
  __shared__ int validL[M];

  const int i = blockIdx.x;
  const int t = threadIdx.x;

  for (int c = t; c < N3; c += blockDim.x) colmax[c] = 0u;
  if (t < M) validL[t] = (nbr[i * KNBR + t] >= 0);

  // build edge features [64 x KPAD]: [x_j (FIN), pos_j - pos_i (3), zero pad]
  for (int e = t; e < M * KPAD; e += blockDim.x) {
    int k = e / KPAD, f = e % KPAD;
    int j  = nbr[i * KNBR + k];
    int jj = (j >= 0) ? j : i;
    float v;
    if (f < FIN)          v = xin[jj * FIN + f];
    else if (f < FIN + 3) v = pos[jj * 3 + (f - FIN)] - pos[i * 3 + (f - FIN)];
    else                  v = 0.0f;
    bufA[e] = (_Float16)v;
  }
  __syncthreads();

  mlp_layer<M, KPAD, N1, true>(bufA, W1p, b1, bufB);
  __syncthreads();
  mlp_layer<M, N1, N2, true>(bufB, W2p, b2, bufA);
  __syncthreads();
  mlp_layer_max<M, N2, N3, true>(bufA, W3p, b3, validL, colmax);
  __syncthreads();

  for (int c = t; c < N3; c += blockDim.x) out[i * N3 + c] = fdec(colmax[c]);
}

// =====================================================================
// Kernel: final per-point MLP [x2(256), pos(3)] -> 1024, fused with
// per-block (32 points) column max. partial[blk*1024 + c].
// =====================================================================
__global__ __launch_bounds__(256) void sa3_kernel(const float* __restrict__ x2,
                                                  const float* __restrict__ pos,
                                                  const unsigned short* W1p, const float* b1,
                                                  const unsigned short* W2p, const float* b2,
                                                  const unsigned short* W3p, const float* b3,
                                                  float* __restrict__ partial) {
  constexpr int M = 32, KPAD = 288, N1 = 256, N2 = 512, N3 = 1024;
  __shared__ __align__(16) _Float16 bufA[M * N2];   // input (288) then layer2 out (512)
  __shared__ __align__(16) _Float16 bufB[M * N1];   // layer1 out (256)
  __shared__ unsigned colmax[N3];

  const int blk = blockIdx.x;   // 0..127 ; points blk*32 .. blk*32+31 (one cloud)
  const int t   = threadIdx.x;

  for (int c = t; c < N3; c += blockDim.x) colmax[c] = 0u;
  for (int e = t; e < M * KPAD; e += blockDim.x) {
    int m = e / KPAD, f = e % KPAD;
    int g = blk * M + m;
    float v;
    if (f < 256)      v = x2[g * 256 + f];
    else if (f < 259) v = pos[g * 3 + (f - 256)];
    else              v = 0.0f;
    bufA[e] = (_Float16)v;
  }
  __syncthreads();

  mlp_layer<M, KPAD, N1, true>(bufA, W1p, b1, bufB);
  __syncthreads();
  mlp_layer<M, N1, N2, true>(bufB, W2p, b2, bufA);
  __syncthreads();
  mlp_layer_max<M, N2, N3, false>(bufA, W3p, b3, nullptr, colmax);
  __syncthreads();

  for (int c = t; c < N3; c += blockDim.x) partial[blk * N3 + c] = fdec(colmax[c]);
}

// Reduce 32 partial blocks per cloud -> out[b, 1024]
__global__ __launch_bounds__(256) void reduce_max_kernel(const float* __restrict__ partial,
                                                         float* __restrict__ out) {
  int b = blockIdx.x;  // 0..3
  for (int c = threadIdx.x; c < 1024; c += blockDim.x) {
    float m = -__builtin_inff();
    for (int k = 0; k < 32; ++k) m = fmaxf(m, partial[(b * 32 + k) * 1024 + c]);
    out[b * 1024 + c] = m;
  }
}

// =====================================================================
// Host launcher
// =====================================================================
extern "C" void kernel_launch(void* const* d_in, const int* in_sizes, int n_in,
                              void* d_out, int out_size, void* d_ws, size_t ws_size,
                              hipStream_t stream) {
  (void)in_sizes; (void)n_in; (void)out_size; (void)ws_size;

  const float* x   = (const float*)d_in[0];
  const float* pos = (const float*)d_in[1];
  // d_in[2] = batch (contiguous equal blocks; unused)
  const float* W[9];
  const float* Bb[9];
  for (int l = 0; l < 9; ++l) {
    W[l]  = (const float*)d_in[3 + 2 * l];
    Bb[l] = (const float*)d_in[3 + 2 * l + 1];
  }

  // workspace carve-up (256B aligned)
  char* w = (char*)d_ws;
  size_t off = 0;
  auto take = [&](size_t bytes) -> void* {
    void* p = w + off;
    off = (off + bytes + 255) & ~(size_t)255;
    return p;
  };
  int*   nbr1    = (int*)take((size_t)NPTS * KNBR * 4);
  int*   nbr2    = (int*)take((size_t)NPTS * KNBR * 4);
  float* x1      = (float*)take((size_t)NPTS * 128 * 4);
  float* x2      = (float*)take((size_t)NPTS * 256 * 4);
  float* partial = (float*)take((size_t)128 * 1024 * 4);

  // packed-weight descriptors: {Kreal, N, KT, NT}
  const int pd[9][4] = {
      {6, 64, 1, 4},   {64, 64, 2, 4},    {64, 128, 2, 8},     // SA1
      {131, 128, 5, 8}, {128, 128, 4, 8}, {128, 256, 4, 16},   // SA2
      {259, 256, 9, 16}, {256, 512, 8, 32}, {512, 1024, 16, 64} // SA3
  };
  unsigned short* Wp[9];
  for (int l = 0; l < 9; ++l) {
    int total = pd[l][2] * pd[l][3] * 512;
    Wp[l] = (unsigned short*)take((size_t)total * 2);
    pack_weights<<<(total + 255) / 256, 256, 0, stream>>>(W[l], Wp[l], pd[l][0],
                                                          pd[l][1], pd[l][3], total);
  }

  // neighbor graphs for r=0.1 and r=0.2
  knn_kernel<<<NPTS, 256, 0, stream>>>(pos, 0.01f, nbr1);
  knn_kernel<<<NPTS, 256, 0, stream>>>(pos, 0.04f, nbr2);

  // SA1: [x_j(3), rel(3)] -> 64 -> 64 -> 128, masked max -> x1
  sa_kernel<3, 32, 64, 64, 128><<<NPTS, 256, 0, stream>>>(
      x, pos, nbr1, Wp[0], Bb[0], Wp[1], Bb[1], Wp[2], Bb[2], x1);

  // SA2: [x1_j(128), rel(3)] -> 128 -> 128 -> 256, masked max -> x2
  sa_kernel<128, 160, 128, 128, 256><<<NPTS, 256, 0, stream>>>(
      x1, pos, nbr2, Wp[3], Bb[3], Wp[4], Bb[4], Wp[5], Bb[5], x2);

  // SA3: [x2(256), pos(3)] -> 256 -> 512 -> 1024, per-block max -> partial
  sa3_kernel<<<128, 256, 0, stream>>>(x2, pos, Wp[6], Bb[6], Wp[7], Bb[7],
                                      Wp[8], Bb[8], partial);

  // global max pool -> out[4,1024]
  reduce_max_kernel<<<NB, 256, 0, stream>>>(partial, (float*)d_out);
}
